// PAMoE_61117384622830
// MI455X (gfx1250) — compile-verified, matrix operand order checked by reading, not run on hardware
//
#include <hip/hip_runtime.h>
#include <math.h>

typedef __bf16 bf16_t;
typedef __attribute__((ext_vector_type(16))) __bf16 v16bf;
typedef __attribute__((ext_vector_type(8)))  __bf16 v8bf;
typedef __attribute__((ext_vector_type(8)))  float  v8f;

#define NUM_EXPERTS 16
#define D_IN  1024
#define D_OUT 512
#define TILE_M 32          // rows per workgroup
#define TILE_K 32          // K per WMMA step

// ------------------------------------------------------------------
// Kernel 0: zero the output accumulator and the per-expert counters
// ------------------------------------------------------------------
__global__ void moe_zero_kernel(float* __restrict__ out, int n, int* __restrict__ counts) {
    int i = blockIdx.x * blockDim.x + threadIdx.x;
    if (i < n) out[i] = 0.0f;
    if (i < NUM_EXPERTS) counts[i] = 0;
}

// ------------------------------------------------------------------
// Kernel T: transpose+convert expert weights once:
//   W[e][k][n] (f32, row-major) -> W_t[e][n][k] (bf16, K contiguous)
// Reads coalesced along n; writes one 16B b128 chunk per thread.
// ------------------------------------------------------------------
__global__ __launch_bounds__(256) void moe_wt_kernel(
    const float* __restrict__ W, bf16_t* __restrict__ Wt)
{
    int t  = blockIdx.x * blockDim.x + threadIdx.x;   // E*512*128 threads
    int n  = t & (D_OUT - 1);                         // 0..511 (fast -> coalesced reads)
    int k0 = ((t >> 9) & 127) * 8;                    // 0..1016 step 8
    int e  = t >> 16;                                 // 512*128 = 65536 per expert
    const float* src = W + ((size_t)e * D_IN + k0) * D_OUT + n;
    v8bf o;
#pragma unroll
    for (int i = 0; i < 8; ++i)
        o[i] = (bf16_t)src[(size_t)i * D_OUT];
    *(v8bf*)(Wt + ((size_t)e * D_OUT + n) * D_IN + k0) = o;
}

// ------------------------------------------------------------------
// Kernel 1: router. One wave32 per row: logits = x_b @ w_router,
// top-2 + softmax over the two logits, atomic dispatch into buckets.
// ------------------------------------------------------------------
__global__ __launch_bounds__(256) void moe_router_kernel(
    const float* __restrict__ x, const float* __restrict__ wr,
    int* __restrict__ counts, int* __restrict__ rowsBuf,
    float* __restrict__ gatesBuf, int B)
{
    const int gw   = (blockIdx.x * blockDim.x + threadIdx.x) >> 5;  // global wave = row
    const int lane = threadIdx.x & 31;
    if (gw >= B) return;

    const float* xr = x + (size_t)gw * D_IN;
    float acc[NUM_EXPERTS];
#pragma unroll
    for (int e = 0; e < NUM_EXPERTS; ++e) acc[e] = 0.0f;

    for (int d = lane; d < D_IN; d += 32) {
        float xv = xr[d];
        const float4* w4 = (const float4*)(wr + (size_t)d * NUM_EXPERTS);
        float4 w0 = w4[0], w1 = w4[1], w2 = w4[2], w3 = w4[3];
        acc[0]  += xv * w0.x;  acc[1]  += xv * w0.y;  acc[2]  += xv * w0.z;  acc[3]  += xv * w0.w;
        acc[4]  += xv * w1.x;  acc[5]  += xv * w1.y;  acc[6]  += xv * w1.z;  acc[7]  += xv * w1.w;
        acc[8]  += xv * w2.x;  acc[9]  += xv * w2.y;  acc[10] += xv * w2.z;  acc[11] += xv * w2.w;
        acc[12] += xv * w3.x;  acc[13] += xv * w3.y;  acc[14] += xv * w3.z;  acc[15] += xv * w3.w;
    }

#pragma unroll
    for (int off = 16; off >= 1; off >>= 1) {
#pragma unroll
        for (int e = 0; e < NUM_EXPERTS; ++e)
            acc[e] += __shfl_xor(acc[e], off, 32);
    }

    if (lane == 0) {
        int   i0 = 0; float v0 = acc[0];
#pragma unroll
        for (int e = 1; e < NUM_EXPERTS; ++e)
            if (acc[e] > v0) { v0 = acc[e]; i0 = e; }
        int   i1 = -1; float v1 = -3.402823466e38f;
#pragma unroll
        for (int e = 0; e < NUM_EXPERTS; ++e)
            if (e != i0 && acc[e] > v1) { v1 = acc[e]; i1 = e; }

        float e1 = expf(v1 - v0);     // softmax over top-2 (v0 >= v1)
        float s  = 1.0f + e1;
        float g0 = 1.0f / s;
        float g1 = e1 / s;

        int p0 = atomicAdd(&counts[i0], 1);
        rowsBuf [i0 * B + p0] = gw;
        gatesBuf[i0 * B + p0] = g0;
        int p1 = atomicAdd(&counts[i1], 1);
        rowsBuf [i1 * B + p1] = gw;
        gatesBuf[i1 * B + p1] = g1;
    }
}

// ------------------------------------------------------------------
// Kernel 2: per-expert gathered GEMM, bf16 WMMA.
// Grid: (B/32 row tiles, 16 experts), block = 256 (8 waves).
// WG computes a 32-row x 512-col block. Wave w: row-half (w>>2),
// col strip (w&3)*128 -> 8 WMMA tiles per wave. A from LDS (staged
// once), B fragments loaded straight from pre-transposed bf16 W_t
// with b128 loads -> no barriers inside the K loop.
// ------------------------------------------------------------------
__global__ __launch_bounds__(256) void moe_expert_gemm_kernel(
    const float* __restrict__ x, const bf16_t* __restrict__ Wt,
    const int* __restrict__ counts, const int* __restrict__ rowsBuf,
    const float* __restrict__ gatesBuf, float* __restrict__ out, int B)
{
    const int e     = blockIdx.y;
    const int cnt   = counts[e];
    const int tile0 = blockIdx.x * TILE_M;
    if (tile0 >= cnt) return;                      // uniform early-exit

    const int tid  = threadIdx.x;
    const int lane = tid & 31;
    const int wave = tid >> 5;
    const int rh   = wave >> 2;                    // row half: rows rh*16..+16
    const int cs   = wave & 3;                     // col strip: cols cs*128..+128

    __shared__ __align__(16) bf16_t As[TILE_M][D_IN];   // 64 KB

    // Stage 32 gathered x rows into LDS as bf16 (float4 coalesced loads).
#pragma unroll 4
    for (int j = 0; j < 32; ++j) {
        int q = j * 256 + tid;            // 0..8191 float4 slots (32 rows * 256)
        int r = q >> 8;                   // row in tile
        int c = (q & 255) * 4;            // column
        float4 v = make_float4(0.f, 0.f, 0.f, 0.f);
        if ((tile0 + r) < cnt) {
            int b = rowsBuf[e * B + tile0 + r];
            v = *(const float4*)(x + (size_t)b * D_IN + c);
        }
        As[r][c + 0] = (bf16_t)v.x;
        As[r][c + 1] = (bf16_t)v.y;
        As[r][c + 2] = (bf16_t)v.z;
        As[r][c + 3] = (bf16_t)v.w;
    }
    __syncthreads();

    const int mrow = lane & 15;           // A row (M) / B col (N) within tile
    const int hi   = lane >> 4;           // half-wave K sub-range select
    const bf16_t* pa0 = &As[rh * 16 + mrow][hi * 8];
    const bf16_t* pb0 = Wt + ((size_t)e * D_OUT + cs * 128 + mrow) * D_IN + hi * 8;

    v8f acc[8];
#pragma unroll
    for (int t = 0; t < 8; ++t) acc[t] = (v8f){0.f,0.f,0.f,0.f,0.f,0.f,0.f,0.f};

    for (int kk = 0; kk < D_IN; kk += TILE_K) {
        // A fragment: elems 0..7 = K[kk+8*hi..], 8..15 = K[kk+16+8*hi..]
        v8bf a_lo = *(const v8bf*)(pa0 + kk);
        v8bf a_hi = *(const v8bf*)(pa0 + kk + 16);
        v16bf afrag = __builtin_shufflevector(a_lo, a_hi,
            0, 1, 2, 3, 4, 5, 6, 7, 8, 9, 10, 11, 12, 13, 14, 15);
#pragma unroll
        for (int t = 0; t < 8; ++t) {     // 8 col tiles -> base + immediate offsets
            const bf16_t* pb = pb0 + (size_t)t * 16 * D_IN + kk;
            v8bf b_lo = *(const v8bf*)pb;
            v8bf b_hi = *(const v8bf*)(pb + 16);
            v16bf bfrag = __builtin_shufflevector(b_lo, b_hi,
                0, 1, 2, 3, 4, 5, 6, 7, 8, 9, 10, 11, 12, 13, 14, 15);
            acc[t] = __builtin_amdgcn_wmma_f32_16x16x32_bf16(
                false, afrag, false, bfrag, (short)0, acc[t], false, false);
        }
    }

    // Epilogue: lane (hi,v) holds M = rh*16 + hi*8 + v, N = cs*128 + t*16 + mrow.
    // combined[b, n] += gate * exp(acc): exactly two addends per output from
    // exact zero -> float atomic order cannot change the result.
#pragma unroll
    for (int v = 0; v < 8; ++v) {
        int m  = rh * 16 + hi * 8 + v;
        int rr = tile0 + m;
        if (rr < cnt) {
            float g = gatesBuf[e * B + rr];
            int   b = rowsBuf [e * B + rr];
            float* op = out + (size_t)b * D_OUT + cs * 128 + mrow;
#pragma unroll
            for (int t = 0; t < 8; ++t)
                atomicAdd(op + t * 16, g * expf(acc[t][v]));
        }
    }
}

// ------------------------------------------------------------------
// Kernel 3: finalize — eps clamp + log, in place.
// ------------------------------------------------------------------
__global__ void moe_finalize_kernel(float* __restrict__ out, int n) {
    int i = blockIdx.x * blockDim.x + threadIdx.x;
    if (i < n) {
        float v = out[i];
        if (v == 0.0f) v = 2.2204460492503131e-16f;  // np.finfo(float).eps
        out[i] = logf(v);
    }
}

// ------------------------------------------------------------------
extern "C" void kernel_launch(void* const* d_in, const int* in_sizes, int n_in,
                              void* d_out, int out_size, void* d_ws, size_t ws_size,
                              hipStream_t stream) {
    const float* x         = (const float*)d_in[0];
    const float* w_router  = (const float*)d_in[1];
    const float* w_experts = (const float*)d_in[2];
    float*       out       = (float*)d_out;

    const int B = in_sizes[0] / D_IN;   // 16384

    // workspace layout
    const size_t EB4   = (size_t)NUM_EXPERTS * B * 4;
    int*    counts   = (int*)d_ws;                                    // 16 ints
    int*    rowsBuf  = (int*)((char*)d_ws + 1024);                    // [E][B] ints
    float*  gatesBuf = (float*)((char*)d_ws + 1024 + EB4);            // [E][B] floats
    bf16_t* Wt       = (bf16_t*)((char*)d_ws + 1024 + 2 * EB4);       // [E][512][1024] bf16

    // 0) zero output accumulator + counters
    moe_zero_kernel<<<(out_size + 255) / 256, 256, 0, stream>>>(out, out_size, counts);

    // T) transpose+convert expert weights to bf16 [e][n][k]
    moe_wt_kernel<<<(NUM_EXPERTS * D_OUT * (D_IN / 8)) / 256, 256, 0, stream>>>(w_experts, Wt);

    // 1) router + dispatch (one wave32 per row)
    moe_router_kernel<<<(B * 32 + 255) / 256, 256, 0, stream>>>(
        x, w_router, counts, rowsBuf, gatesBuf, B);

    // 2) per-expert WMMA GEMM + gate*exp combine
    dim3 grid((B + TILE_M - 1) / TILE_M, NUM_EXPERTS);
    moe_expert_gemm_kernel<<<grid, 256, 0, stream>>>(
        x, Wt, counts, rowsBuf, gatesBuf, out, B);

    // 3) eps clamp + log
    moe_finalize_kernel<<<(out_size + 255) / 256, 256, 0, stream>>>(out, out_size);
}
